// GraphSAGEContrastive_48318382080416
// MI455X (gfx1250) — matrix operand order repaired and checked by reading, not compiled, not run on hardware
//
#include <hip/hip_runtime.h>
#include <hip/hip_bf16.h>

typedef __bf16 bf16_t;
typedef __bf16    v16bf __attribute__((ext_vector_type(16)));
typedef float     v8f   __attribute__((ext_vector_type(8)));
typedef unsigned  uv4   __attribute__((ext_vector_type(4)));

union FragV { v16bf v; uv4 u[2]; };

#define IN_DIM 300
#define HID    256
#define EMB    128
#define PROJ   64

// 1 = stage B into LDS with CDNA5 async-to-LDS (ASYNCcnt) path; 0 = VGPR bounce
#define USE_ASYNC_STAGE 1

// ---------------------------------------------------------------------------
// Degree accumulation: deg[dst[e]] += 1
// ---------------------------------------------------------------------------
__global__ void deg_kernel(const long long* __restrict__ dst, float* __restrict__ deg,
                           long long E) {
    long long e = (long long)blockIdx.x * blockDim.x + threadIdx.x;
    if (e >= E) return;
    atomicAdd(&deg[dst[e]], 1.0f);
}

__global__ void invdeg_kernel(const float* __restrict__ deg, float* __restrict__ invdeg,
                              int M, int Mp) {
    int i = blockIdx.x * blockDim.x + threadIdx.x;
    if (i >= Mp) return;
    invdeg[i] = (i < M) ? 1.0f / fmaxf(deg[i], 1.0f) : 0.0f;
}

// ---------------------------------------------------------------------------
// Convert f32 rows -> padded bf16 rows (optional per-row scale), zero padding
// ---------------------------------------------------------------------------
__global__ void cvt_rows_kernel(const float* __restrict__ S, int ldS, int K,
                                const float* __restrict__ rowScale,
                                bf16_t* __restrict__ D, int ldD, int colOff, int Kpad,
                                int M, long long total) {
    long long gid = (long long)blockIdx.x * blockDim.x + threadIdx.x;
    if (gid >= total) return;
    long long m = gid / Kpad;
    int k = (int)(gid % Kpad);
    float v = 0.0f;
    if (m < M && k < K) {
        v = S[m * (long long)ldS + k];
        if (rowScale) v *= rowScale[m];
    }
    D[m * (long long)ldD + colOff + k] = (bf16_t)v;
}

// ---------------------------------------------------------------------------
// Weight transpose + bf16: Wt[n][kOff+k] = W[k][n] (k<K), zero-pad to Kpad
// ---------------------------------------------------------------------------
__global__ void build_wt_kernel(const float* __restrict__ W, int K, int N, int Kpad,
                                bf16_t* __restrict__ Wt, int rowStride, int kOff) {
    long long gid = (long long)blockIdx.x * blockDim.x + threadIdx.x;
    long long total = (long long)N * Kpad;
    if (gid >= total) return;
    int n = (int)(gid / Kpad);
    int k = (int)(gid % Kpad);
    float v = (k < K) ? W[(long long)k * N + n] : 0.0f;
    Wt[(long long)n * rowStride + kOff + k] = (bf16_t)v;
}

// ---------------------------------------------------------------------------
// Fold eval-mode BN + bias into per-channel scale/shift:
//   y = gemm*s + t,  s = g*rsqrt(v+eps),  t = (b - mean)*s + beta
// ---------------------------------------------------------------------------
__global__ void bn_prep_kernel(const float* g, const float* beta, const float* mean,
                               const float* var, const float* bias,
                               float* s, float* t, int n) {
    int i = blockIdx.x * blockDim.x + threadIdx.x;
    if (i >= n) return;
    float sc = g[i] * rsqrtf(var[i] + 1e-5f);
    s[i] = sc;
    t[i] = (bias[i] - mean[i]) * sc + beta[i];
}

// ---------------------------------------------------------------------------
// Edge scatter: agg[dst] += bf16(X[src])  (f32 atomics, 8B bf16 gathers)
// ---------------------------------------------------------------------------
__global__ void scatter_kernel(const bf16_t* __restrict__ Xb, int ldx,
                               const long long* __restrict__ src,
                               const long long* __restrict__ dst,
                               float* __restrict__ agg, int ldagg,
                               int nch, long long E) {
    long long e = (long long)blockIdx.x * blockDim.x + threadIdx.x;
    if (e >= E) return;
    long long s = src[e], d = dst[e];
    const bf16_t* p = Xb + s * (long long)ldx;
    float* q = agg + d * (long long)ldagg;
    for (int c = 0; c < nch; ++c) {
        int o = c * 4;
        union { unsigned long long w; bf16_t h[4]; } u;
        u.w = *(const unsigned long long*)(p + o);
        atomicAdd(q + o + 0, (float)u.h[0]);
        atomicAdd(q + o + 1, (float)u.h[1]);
        atomicAdd(q + o + 2, (float)u.h[2]);
        atomicAdd(q + o + 3, (float)u.h[3]);
    }
}

// ---------------------------------------------------------------------------
// bf16 WMMA GEMM: C[M][N] = A[Mp][Kp] @ Wt[N][Kp]^T.
// B (weights, 64-col stripe) staged once into LDS via async-to-LDS, read back
// with ds_load_b128 (row stride Kp+8 -> bank-conflict-free). Inner loop is
// branch-free and double-buffers BOTH A (global) and B (LDS) fragments,
// advancing 2 k-steps per iteration (all Kp used have an even step count).
// Epilogue: scale/shift/relu, f32+bf16 out.
// Wave computes 16 rows x 64 cols; block = 8 waves = 128 rows.
// Grid = (Mp/128, N/64). Dynamic LDS = 64*(Kp+8)*2 + 64 bytes.
// ---------------------------------------------------------------------------
__global__ __launch_bounds__(256)
void wmma_gemm_kernel(const bf16_t* __restrict__ A, int lda,
                      const bf16_t* __restrict__ Wt, int Kp,
                      const float* __restrict__ scale,
                      const float* __restrict__ shift, int relu,
                      float* __restrict__ Cf, int ldcf,
                      bf16_t* __restrict__ Cb, int ldcb, int cbOff,
                      int M) {
    extern __shared__ __align__(16) char smem_raw[];
    bf16_t* sB = (bf16_t*)smem_raw;          // [64][KpL] (+64B slack for load-ahead)
    const int KpL = Kp + 8;                  // pad: 16 lanes spread over 64 banks

    const int lane = threadIdx.x & 31;
    const int wave = threadIdx.x >> 5;
    const int half = lane >> 4;   // 0 or 1
    const int l16  = lane & 15;
    const int m0 = blockIdx.x * 128 + wave * 16;
    const int n0 = blockIdx.y * 64;

    // ---- cooperative stage of the 64-column weight stripe into LDS ----
    {
        const int chunksPerRow = Kp >> 3;            // b128 = 8 bf16
        const int totalChunks  = 64 * chunksPerRow;  // multiple of 256 for all Kp used
        for (int c = threadIdx.x; c < totalChunks; c += 256) {
            int n = c / chunksPerRow;
            int j = c - n * chunksPerRow;
            const bf16_t* g = Wt + (size_t)(n0 + n) * Kp + j * 8;
            bf16_t*       l = sB + (size_t)n * KpL + j * 8;
#if USE_ASYNC_STAGE
            unsigned loff = (unsigned)(unsigned long long)l;   // LDS byte offset
            asm volatile("global_load_async_to_lds_b128 %0, %1, off"
                         :: "v"(loff), "v"(g) : "memory");
#else
            *(uv4*)l = *(const uv4*)g;
#endif
        }
#if USE_ASYNC_STAGE
        asm volatile("s_wait_asynccnt 0" ::: "memory");
#endif
        __syncthreads();
    }

    const bf16_t* arow = A + (size_t)(m0 + l16) * lda + half * 8;
    const bf16_t* brow = sB + (size_t)l16 * KpL + half * 8;

    v8f acc[4];
    acc[0] = (v8f)0.0f; acc[1] = (v8f)0.0f; acc[2] = (v8f)0.0f; acc[3] = (v8f)0.0f;

    // A fragment (16x32 bf16): lanes 0-15 -> K k..k+7 / k+16..k+23, lanes 16-31 +8
    auto loadA = [&](FragV& f, int k) {
        f.u[0] = *(const uv4*)(arow + k);
        f.u[1] = *(const uv4*)(arow + k + 16);
    };
    // B fragment (32x16): col = n, lanes 0-15 K k..k+7 / k+16..k+23, lanes 16-31 +8
    auto loadB = [&](FragV* f, int k) {
#pragma unroll
        for (int nt = 0; nt < 4; ++nt) {
            const bf16_t* bp = brow + (size_t)(nt * 16) * KpL + k;
            f[nt].u[0] = *(const uv4*)(bp);
            f[nt].u[1] = *(const uv4*)(bp + 16);
        }
    };

    FragV aA, aB;
    FragV bA[4], bB[4];
    loadA(aA, 0);
    loadB(bA, 0);

    // Branch-free ping-pong; final load-ahead over-reads <=1 k-step (A stays
    // inside d_ws, LDS OOB reads return 0) and its data is never used.
    for (int k = 0; k < Kp; k += 64) {
        loadA(aB, k + 32);
        loadB(bB, k + 32);
        __builtin_prefetch(arow + k + 128, 0, 1);
#pragma unroll
        for (int nt = 0; nt < 4; ++nt)
            acc[nt] = __builtin_amdgcn_wmma_f32_16x16x32_bf16(
                false, aA.v, false, bA[nt].v, (short)0, acc[nt], false, false);
        loadA(aA, k + 64);
        loadB(bA, k + 64);
#pragma unroll
        for (int nt = 0; nt < 4; ++nt)
            acc[nt] = __builtin_amdgcn_wmma_f32_16x16x32_bf16(
                false, aB.v, false, bB[nt].v, (short)0, acc[nt], false, false);
    }

#pragma unroll
    for (int nt = 0; nt < 4; ++nt) {
        int n = n0 + nt * 16 + l16;
        float sc = scale ? scale[n] : 1.0f;
        float sh = shift ? shift[n] : 0.0f;
#pragma unroll
        for (int r = 0; r < 8; ++r) {
            int m = m0 + r + 8 * half;   // C layout: lanes 0-15 rows r, 16-31 rows r+8
            float v = acc[nt][r] * sc + sh;
            if (relu) v = fmaxf(v, 0.0f);
            bool in = (m < M);
            if (Cf && in) Cf[(size_t)m * ldcf + n] = v;
            if (Cb) Cb[(size_t)m * ldcb + cbOff + n] = (bf16_t)(in ? v : 0.0f);
        }
    }
}

// ---------------------------------------------------------------------------
extern "C" void kernel_launch(void* const* d_in, const int* in_sizes, int n_in,
                              void* d_out, int out_size, void* d_ws, size_t ws_size,
                              hipStream_t stream) {
    const float*     x    = (const float*)d_in[0];
    const long long* ei   = (const long long*)d_in[1];
    const float*     W1l  = (const float*)d_in[2];
    const float*     b1   = (const float*)d_in[3];
    const float*     W1r  = (const float*)d_in[4];
    const float*     bn_g = (const float*)d_in[5];
    const float*     bn_b = (const float*)d_in[6];
    const float*     bn_m = (const float*)d_in[7];
    const float*     bn_v = (const float*)d_in[8];
    const float*     W2l  = (const float*)d_in[9];
    const float*     b2   = (const float*)d_in[10];
    const float*     W2r  = (const float*)d_in[11];
    const float*     Wp1  = (const float*)d_in[12];
    const float*     bp1  = (const float*)d_in[13];
    const float*     Wp2  = (const float*)d_in[14];
    const float*     bp2  = (const float*)d_in[15];
    float*           out  = (float*)d_out;

    const int       M  = in_sizes[0] / IN_DIM;       // 100000
    const long long E  = in_sizes[1] / 2;            // 1600000
    const long long* src = ei;
    const long long* dst = ei + E;
    const int Mp = ((M + 127) / 128) * 128;

    // ---- workspace layout (256B aligned) ----
    char* ws = (char*)d_ws;
    size_t o = 0;
    auto walloc = [&](size_t bytes) {
        void* p = ws + o;
        o = (o + bytes + 255) & ~(size_t)255;
        return p;
    };
    float*  invdeg = (float*)walloc((size_t)Mp * 4);
    float*  deg    = (float*)walloc((size_t)Mp * 4);
    bf16_t* Ab1    = (bf16_t*)walloc((size_t)Mp * 640 * 2);  // [Mp][320 agg | 320 x]
    float*  aggf   = (float*)walloc((size_t)Mp * IN_DIM * 4);
    bf16_t* Ab2    = (bf16_t*)walloc((size_t)Mp * 512 * 2);  // [Mp][256 agg2 | 256 h]
    bf16_t* Wc1    = (bf16_t*)walloc((size_t)HID * 640 * 2);
    bf16_t* Wc2    = (bf16_t*)walloc((size_t)EMB * 512 * 2);
    bf16_t* Wp1t   = (bf16_t*)walloc((size_t)(HID/2) * 128 * 2);
    bf16_t* Wp2t   = (bf16_t*)walloc((size_t)PROJ * 128 * 2);
    float*  s1     = (float*)walloc(HID * 4);
    float*  t1     = (float*)walloc(HID * 4);
    bf16_t* embb   = Ab1;                        // reuse (Ab1 dead after gemm1)
    bf16_t* hidb   = Ab1 + (size_t)Mp * 128;     // disjoint from embb
    (void)n_in; (void)out_size; (void)ws_size;

    const int T = 256;
    auto blks = [&](long long n) { return (unsigned)((n + T - 1) / T); };
    auto ldsB = [](int Kp) { return (size_t)64 * (Kp + 8) * 2 + 64; };

    // 1) degrees
    hipMemsetAsync(deg, 0, (size_t)Mp * 4, stream);
    deg_kernel<<<blks(E), T, 0, stream>>>(dst, deg, E);
    invdeg_kernel<<<blks(Mp), T, 0, stream>>>(deg, invdeg, M, Mp);

    // 2) x -> bf16 into Ab1 cols [320,640)
    long long tot320 = (long long)Mp * 320;
    cvt_rows_kernel<<<blks(tot320), T, 0, stream>>>(x, IN_DIM, IN_DIM, nullptr,
                                                    Ab1, 640, 320, 320, M, tot320);
    // 3) scatter layer 1 (gather bf16 x, f32 atomic accumulate)
    hipMemsetAsync(aggf, 0, (size_t)Mp * IN_DIM * 4, stream);
    scatter_kernel<<<blks(E), T, 0, stream>>>(Ab1 + 320, 640, src, dst,
                                              aggf, IN_DIM, IN_DIM / 4, E);
    // 4) agg*inv_deg -> bf16 into Ab1 cols [0,320)
    cvt_rows_kernel<<<blks(tot320), T, 0, stream>>>(aggf, IN_DIM, IN_DIM, invdeg,
                                                    Ab1, 640, 0, 320, M, tot320);
    // 5) weights + BN fold
    build_wt_kernel<<<blks((long long)HID * 320), T, 0, stream>>>(W1l, IN_DIM, HID, 320, Wc1, 640, 0);
    build_wt_kernel<<<blks((long long)HID * 320), T, 0, stream>>>(W1r, IN_DIM, HID, 320, Wc1, 640, 320);
    bn_prep_kernel<<<1, 256, 0, stream>>>(bn_g, bn_b, bn_m, bn_v, b1, s1, t1, HID);

    // 6) GEMM1: h = relu(BN(Ab1 @ Wc1^T)) -> bf16 into Ab2 cols [256,512)
    wmma_gemm_kernel<<<dim3(Mp / 128, HID / 64), T, ldsB(640), stream>>>(
        Ab1, 640, Wc1, 640, s1, t1, 1, nullptr, 0, Ab2, 512, 256, M);

    // 7) scatter layer 2 on h (bf16)
    hipMemsetAsync(aggf, 0, (size_t)Mp * HID * 4, stream);
    scatter_kernel<<<blks(E), T, 0, stream>>>(Ab2 + 256, 512, src, dst,
                                              aggf, HID, HID / 4, E);
    // 8) agg2*inv_deg -> bf16 into Ab2 cols [0,256)
    long long tot256 = (long long)Mp * 256;
    cvt_rows_kernel<<<blks(tot256), T, 0, stream>>>(aggf, HID, HID, invdeg,
                                                    Ab2, 512, 0, 256, M, tot256);
    // 9) GEMM2: emb = Ab2 @ Wc2^T + b2 -> f32 d_out + bf16 embb
    build_wt_kernel<<<blks((long long)EMB * 256), T, 0, stream>>>(W2l, HID, EMB, 256, Wc2, 512, 0);
    build_wt_kernel<<<blks((long long)EMB * 256), T, 0, stream>>>(W2r, HID, EMB, 256, Wc2, 512, 256);
    wmma_gemm_kernel<<<dim3(Mp / 128, EMB / 64), T, ldsB(512), stream>>>(
        Ab2, 512, Wc2, 512, nullptr, b2, 0, out, EMB, embb, EMB, 0, M);

    // 10) projection head
    build_wt_kernel<<<blks((long long)(HID/2) * 128), T, 0, stream>>>(Wp1, EMB, HID/2, 128, Wp1t, 128, 0);
    build_wt_kernel<<<blks((long long)PROJ * 128), T, 0, stream>>>(Wp2, HID/2, PROJ, 128, Wp2t, 128, 0);
    wmma_gemm_kernel<<<dim3(Mp / 128, (HID/2) / 64), T, ldsB(128), stream>>>(
        embb, 128, Wp1t, 128, nullptr, bp1, 1, nullptr, 0, hidb, 128, 0, M);
    wmma_gemm_kernel<<<dim3(Mp / 128, PROJ / 64), T, ldsB(128), stream>>>(
        hidb, 128, Wp2t, 128, nullptr, bp2, 0, out + (size_t)M * EMB, PROJ, nullptr, 0, 0, M);
}
